// AuxiliaryMultiHeadedAttention_23347442221588
// MI455X (gfx1250) — matrix operand order repaired and checked by reading, compiled, not run on hardware
//
#include <hip/hip_runtime.h>

// ---------------------------------------------------------------------------
// AuxiliaryMultiHeadedAttention for MI455X (gfx1250, wave32, WMMA bf16)
//   B=4, S=1024, HID=1024, H=16, DH=64
// Phase 0: one-time f32 -> bf16 conversion of activations + weights (ws).
// Phase 1: Q/K'/V projections (bf16 GEMMs, ping-pong pipelined fragments),
//          K' = k@Wk^T + k_b@Wkb^T, scale folded into Q, V stored [dh][s].
// Phase 2: flash attention, scores transposed (keys x queries), online
//          softmax in registers, ping-pong prefetched K' fragments.
// Phase 3: hidden @ Wo^T + bo -> f32 out (ping-pong pipelined).
// ---------------------------------------------------------------------------

typedef __attribute__((ext_vector_type(16))) __bf16 v16bf;
typedef __attribute__((ext_vector_type(8)))  float  v8f;
typedef unsigned short u16;

#define B_    4
#define S_    1024
#define HID_  1024
#define H_    16
#define DH_   64
#define SCALE_ 0.07216878364870323f   /* 1/sqrt(3*DH) = 1/sqrt(192) */

__device__ __forceinline__ u16 f2bf(float f) {
  unsigned u = __float_as_uint(f);
  u += 0x7FFFu + ((u >> 16) & 1u);    // round to nearest even
  return (u16)(u >> 16);
}

__device__ __forceinline__ v8f wmma_bf16(v16bf a, v16bf b, v8f c) {
  return __builtin_amdgcn_wmma_f32_16x16x32_bf16(false, a, false, b, (short)0, c,
                                                 false, false);
}

// --- A fragment 16x32 (M x K), bf16, from row-major [rows x ld] source.
__device__ __forceinline__ v16bf load_a_bf16(const u16* __restrict__ base, int ld,
                                             int row0, int k0) {
  const int lane = threadIdx.x & 31;
  const int r  = row0 + (lane & 15);
  const int hi = lane >> 4;
  const u16* p = base + (size_t)r * ld + k0 + hi * 16;
  union { v16bf v; uint4 q[2]; } f;
  f.q[0] = *reinterpret_cast<const uint4*>(p);
  f.q[1] = *reinterpret_cast<const uint4*>(p + 8);
  return f.v;
}

// --- B fragment 32x16 (K x N), bf16, source is B^T row-major [N x ld].
__device__ __forceinline__ v16bf load_b_bf16(const u16* __restrict__ base, int ld,
                                             int n0, int k0) {
  const int lane = threadIdx.x & 31;
  const int n  = n0 + (lane & 15);
  const int hi = lane >> 4;
  const u16* p = base + (size_t)n * ld + k0 + hi * 16;
  union { v16bf v; uint4 q[2]; } f;
  f.q[0] = *reinterpret_cast<const uint4*>(p);
  f.q[1] = *reinterpret_cast<const uint4*>(p + 8);
  return f.v;
}

// ---------------------------------------------------------------------------
// Phase 0: bulk f32 -> bf16 conversion. 8 elements / thread, 16B stores.
// ---------------------------------------------------------------------------
struct CvtArgs {
  const float* src[9];
  u16*         dst[9];
  int          n[9];
};

__global__ __launch_bounds__(256) void cvt_kernel(CvtArgs args) {
  const int z = blockIdx.z;
  const float* __restrict__ s = args.src[z];
  u16* __restrict__ d = args.dst[z];
  const int n  = args.n[z];
  const int i0 = (blockIdx.x * 256 + threadIdx.x) * 8;
  if (i0 >= n) return;
  const float4 f0 = *reinterpret_cast<const float4*>(s + i0);
  const float4 f1 = *reinterpret_cast<const float4*>(s + i0 + 4);
  union { uint4 x; u16 u[8]; } o;
  o.u[0] = f2bf(f0.x); o.u[1] = f2bf(f0.y); o.u[2] = f2bf(f0.z); o.u[3] = f2bf(f0.w);
  o.u[4] = f2bf(f1.x); o.u[5] = f2bf(f1.y); o.u[6] = f2bf(f1.z); o.u[7] = f2bf(f1.w);
  *reinterpret_cast<uint4*>(d + i0) = o.x;
}

// ---------------------------------------------------------------------------
// Ping-pong pipelined GEMM strip: acc[4] += A(m0,:) * B(n0..n0+63,:)^T.
// Even/odd buffers are loaded in place by fixed load sites -> no register
// rotation copies; loads of one phase overlap the other phase's WMMAs.
// ---------------------------------------------------------------------------
__device__ __forceinline__ void gemm_strip(const u16* __restrict__ X,
                                           const u16* __restrict__ W,
                                           int m0, int n0, v8f acc[4]) {
  v16bf aE = load_a_bf16(X, HID_, m0, 0);
  v16bf bE[4];
#pragma unroll
  for (int t = 0; t < 4; ++t) bE[t] = load_b_bf16(W, HID_, n0 + 16 * t, 0);
#pragma unroll 1
  for (int kk0 = 0; kk0 < HID_; kk0 += 64) {
    const int k1 = kk0 + 32;                 // always < HID_
    v16bf aO = load_a_bf16(X, HID_, m0, k1);
    v16bf bO[4];
#pragma unroll
    for (int t = 0; t < 4; ++t) bO[t] = load_b_bf16(W, HID_, n0 + 16 * t, k1);
#pragma unroll
    for (int t = 0; t < 4; ++t) acc[t] = wmma_bf16(aE, bE[t], acc[t]);
    const int k2 = (kk0 + 64) & (HID_ - 1);  // wraps to 0 on last iter (dead)
    aE = load_a_bf16(X, HID_, m0, k2);
#pragma unroll
    for (int t = 0; t < 4; ++t) bE[t] = load_b_bf16(W, HID_, n0 + 16 * t, k2);
#pragma unroll
    for (int t = 0; t < 4; ++t) acc[t] = wmma_bf16(aO, bO[t], acc[t]);
  }
}

// ---------------------------------------------------------------------------
// Phase 1: projections. blockIdx.z: 0 -> Q (scaled), 1 -> K' = K+Kb, 2 -> V^T.
// One wave per 16x64 strip. Grid (32, 16, 3), block 256 (8 waves).
// ---------------------------------------------------------------------------
__global__ __launch_bounds__(256) void proj_kernel(
    const u16* __restrict__ qbf,  const u16* __restrict__ kbf,
    const u16* __restrict__ vbf,  const u16* __restrict__ kbbf,
    const u16* __restrict__ Wqbf, const u16* __restrict__ Wkbf,
    const u16* __restrict__ Wvbf, const u16* __restrict__ Wkbbf,
    const float* __restrict__ bq, const float* __restrict__ bk,
    const float* __restrict__ bv, const float* __restrict__ bkb,
    u16* __restrict__ qh, u16* __restrict__ kkp, u16* __restrict__ vt) {
  const int wave  = threadIdx.x >> 5;
  const int m0    = (blockIdx.x * 8 + wave) * 16;
  const int n0    = blockIdx.y * 64;
  const int mode  = blockIdx.z;
  const int lane  = threadIdx.x & 31;
  const int hi    = lane >> 4;
  const int qsel  = lane & 15;
  const v8f zero8 = {0.f, 0.f, 0.f, 0.f, 0.f, 0.f, 0.f, 0.f};
  v8f acc[4] = {zero8, zero8, zero8, zero8};

  if (mode == 0) {                                   // Q projection (scaled)
    gemm_strip(qbf, Wqbf, m0, n0, acc);
#pragma unroll
    for (int t = 0; t < 4; ++t) {
      const int col = n0 + 16 * t + qsel;
      const int h = col >> 6, dh = col & (DH_ - 1);
      const float bias = bq[col];
#pragma unroll
      for (int j = 0; j < 8; ++j) {
        const int m = m0 + j + 8 * hi;
        const int bi = m >> 10, s = m & (S_ - 1);
        qh[(((size_t)(bi * H_ + h)) * S_ + s) * DH_ + dh] =
            f2bf((acc[t][j] + bias) * SCALE_);
      }
    }
  } else if (mode == 1) {
    // K' ping-pong: even phase always (k, Wk), odd phase always (k_b, Wkb).
    // Fixed pointers per load site -> global_load path, no rotation copies.
    v16bf aA = load_a_bf16(kbf, HID_, m0, 0);
    v16bf bA[4];
#pragma unroll
    for (int t = 0; t < 4; ++t) bA[t] = load_b_bf16(Wkbf, HID_, n0 + 16 * t, 0);
#pragma unroll 1
    for (int i = 0; i < 32; ++i) {
      const int kk0 = i * 32;
      v16bf aB = load_a_bf16(kbbf, HID_, m0, kk0);
      v16bf bB[4];
#pragma unroll
      for (int t = 0; t < 4; ++t) bB[t] = load_b_bf16(Wkbbf, HID_, n0 + 16 * t, kk0);
#pragma unroll
      for (int t = 0; t < 4; ++t) acc[t] = wmma_bf16(aA, bA[t], acc[t]);
      const int kn = (kk0 + 32) & (HID_ - 1);        // wraps on last iter (dead)
      aA = load_a_bf16(kbf, HID_, m0, kn);
#pragma unroll
      for (int t = 0; t < 4; ++t) bA[t] = load_b_bf16(Wkbf, HID_, n0 + 16 * t, kn);
#pragma unroll
      for (int t = 0; t < 4; ++t) acc[t] = wmma_bf16(aB, bB[t], acc[t]);
    }
#pragma unroll
    for (int t = 0; t < 4; ++t) {
      const int col = n0 + 16 * t + qsel;
      const int h = col >> 6, dh = col & (DH_ - 1);
      const float bias = bk[col] + bkb[col];
#pragma unroll
      for (int j = 0; j < 8; ++j) {
        const int m = m0 + j + 8 * hi;
        const int bi = m >> 10, s = m & (S_ - 1);
        kkp[(((size_t)(bi * H_ + h)) * S_ + s) * DH_ + dh] = f2bf(acc[t][j] + bias);
      }
    }
  } else {                                           // V projection, stored [dh][s]
    gemm_strip(vbf, Wvbf, m0, n0, acc);
#pragma unroll
    for (int t = 0; t < 4; ++t) {
      const int col = n0 + 16 * t + qsel;
      const int h = col >> 6, dh = col & (DH_ - 1);
      const float bias = bv[col];
#pragma unroll
      for (int j = 0; j < 8; ++j) {
        const int m = m0 + j + 8 * hi;
        const int bi = m >> 10, s = m & (S_ - 1);
        vt[(((size_t)(bi * H_ + h)) * DH_ + dh) * S_ + s] = f2bf(acc[t][j] + bias);
      }
    }
  }
}

// ---------------------------------------------------------------------------
// Phase 2: flash attention. One wave per (b, h, 16-query tile): 4096 waves.
// Chunk loop unrolled x2 with even/odd K'/mask buffers (no rotation copies).
// ---------------------------------------------------------------------------
__global__ __launch_bounds__(256) void attn_kernel(
    const u16* __restrict__ qh, const u16* __restrict__ kkp,
    const u16* __restrict__ vt, const int* __restrict__ mask,
    u16* __restrict__ hid) {
  const int gw    = blockIdx.x * 8 + (threadIdx.x >> 5);
  const int b     = gw >> 10;          // H_*S_/16 = 1024 tiles per batch
  const int rem   = gw & 1023;
  const int h     = rem >> 6;
  const int q0    = (rem & 63) * 16;
  const int lane  = threadIdx.x & 31;
  const int qsel  = lane & 15;
  const int hi    = lane >> 4;
  const u16* qhd  = qh  + ((size_t)(b * H_ + h)) * S_ * DH_;
  const u16* kkd  = kkp + ((size_t)(b * H_ + h)) * S_ * DH_;
  const u16* vtd  = vt  + ((size_t)(b * H_ + h)) * DH_ * S_;
  const int* mp   = mask + b * S_;
  const v8f zero8 = {0.f, 0.f, 0.f, 0.f, 0.f, 0.f, 0.f, 0.f};

  // Q B-fragments are loop-invariant (K-dim = DH = 64 -> two 32-chunks).
  const v16bf qf0 = load_b_bf16(qhd, DH_, q0, 0);
  const v16bf qf1 = load_b_bf16(qhd, DH_, q0, 32);

  v8f o[4] = {zero8, zero8, zero8, zero8};   // O^T: 4 tiles of (16 dh x 16 q)
  float mrun = -1e30f, lrun = 0.f;

  // process one 32-key chunk given its K' fragments + mask bias value
  auto process_chunk = [&](int kb0c, const v16bf* kcur, float fbcur) {
    v8f st[2];
#pragma unroll
    for (int t = 0; t < 2; ++t) {
      v8f c = zero8;
      c = wmma_bf16(kcur[2 * t],     qf0, c);
      c = wmma_bf16(kcur[2 * t + 1], qf1, c);
#pragma unroll
      for (int j = 0; j < 8; ++j)           // key = kb0c + 16t + j + 8*hi
        c[j] += __shfl(fbcur, 16 * t + j + (hi << 3), 32);
      st[t] = c;
    }
    float cmax = -3.0e38f;
#pragma unroll
    for (int t = 0; t < 2; ++t)
#pragma unroll
      for (int j = 0; j < 8; ++j) cmax = fmaxf(cmax, st[t][j]);
    cmax = fmaxf(cmax, __shfl_xor(cmax, 16, 32));
    const float mnew  = fmaxf(mrun, cmax);
    const float alpha = __expf(mrun - mnew);

    // issue V fragment loads now; repack VALU below hides their latency
    v16bf av[4];
#pragma unroll
    for (int d = 0; d < 4; ++d) av[d] = load_a_bf16(vtd, S_, 16 * d, kb0c);

    float csum = 0.f;
#pragma unroll
    for (int t = 0; t < 2; ++t)
#pragma unroll
      for (int j = 0; j < 8; ++j) {
        const float p = __expf(st[t][j] - mnew);
        st[t][j] = p;
        csum += p;
      }
    csum += __shfl_xor(csum, 16, 32);
    lrun = lrun * alpha + csum;
    mrun = mnew;
#pragma unroll
    for (int d = 0; d < 4; ++d)
#pragma unroll
      for (int j = 0; j < 8; ++j) o[d][j] *= alpha;  // alpha lane-uniform per query

    union { v16bf v; u16 u[16]; } pf;                 // repack P^T -> B fragment
#pragma unroll
    for (int i = 0; i < 8; ++i) {
      const float s0a = __shfl(st[0][i], qsel,      32);
      const float s1a = __shfl(st[1][i], qsel,      32);
      const float s0b = __shfl(st[0][i], qsel + 16, 32);
      const float s1b = __shfl(st[1][i], qsel + 16, 32);
      pf.u[i]     = f2bf(hi ? s1a : s0a);
      pf.u[8 + i] = f2bf(hi ? s1b : s0b);
    }
#pragma unroll
    for (int d = 0; d < 4; ++d) o[d] = wmma_bf16(av[d], pf.v, o[d]);
  };

  // even-phase K' fragments + mask for chunk 0
  v16bf kE[4];
  kE[0] = load_a_bf16(kkd, DH_, 0, 0);
  kE[1] = load_a_bf16(kkd, DH_, 0, 32);
  kE[2] = load_a_bf16(kkd, DH_, 16, 0);
  kE[3] = load_a_bf16(kkd, DH_, 16, 32);
  float fbE = (mp[lane] == 0) ? -1e9f : 0.f;

#pragma unroll 1
  for (int kb0 = 0; kb0 < S_; kb0 += 64) {
    // prefetch odd chunk (kb0+32, always < S_)
    const int ob = kb0 + 32;
    v16bf kO[4];
    kO[0] = load_a_bf16(kkd, DH_, ob, 0);
    kO[1] = load_a_bf16(kkd, DH_, ob, 32);
    kO[2] = load_a_bf16(kkd, DH_, ob + 16, 0);
    kO[3] = load_a_bf16(kkd, DH_, ob + 16, 32);
    const float fbO = (mp[ob + lane] == 0) ? -1e9f : 0.f;

    process_chunk(kb0, kE, fbE);

    // prefetch next even chunk (wraps to 0 on last iter; values dead)
    const int nb = (kb0 + 64) & (S_ - 1);
    kE[0] = load_a_bf16(kkd, DH_, nb, 0);
    kE[1] = load_a_bf16(kkd, DH_, nb, 32);
    kE[2] = load_a_bf16(kkd, DH_, nb + 16, 0);
    kE[3] = load_a_bf16(kkd, DH_, nb + 16, 32);
    fbE = (mp[nb + lane] == 0) ? -1e9f : 0.f;

    process_chunk(ob, kO, fbO);
  }

  // normalize + store hidden bf16 [B*S, HID]; dh = 16d + 8*hi + j -> 16B stores
  const float linv = 1.f / lrun;
  const size_t rowbase = ((size_t)(b * S_ + q0 + qsel)) * HID_ + h * DH_;
#pragma unroll
  for (int d = 0; d < 4; ++d) {
    union { uint4 x; u16 u[8]; } s8;
#pragma unroll
    for (int j = 0; j < 8; ++j) s8.u[j] = f2bf(o[d][j] * linv);
    *reinterpret_cast<uint4*>(hid + rowbase + 16 * d + 8 * hi) = s8.x;
  }
}

// ---------------------------------------------------------------------------
// Phase 3: out = hidden @ Wo^T + bo (f32 out). 16x64 strip/wave. Grid (32, 16).
// ---------------------------------------------------------------------------
__global__ __launch_bounds__(256) void outproj_kernel(
    const u16* __restrict__ hid, const u16* __restrict__ Wobf,
    const float* __restrict__ bo, float* __restrict__ out) {
  const int wave = threadIdx.x >> 5;
  const int m0   = (blockIdx.x * 8 + wave) * 16;
  const int n0   = blockIdx.y * 64;
  const int lane = threadIdx.x & 31;
  const int hi   = lane >> 4;
  const int qsel = lane & 15;
  const v8f zero8 = {0.f, 0.f, 0.f, 0.f, 0.f, 0.f, 0.f, 0.f};
  v8f acc[4] = {zero8, zero8, zero8, zero8};
  gemm_strip(hid, Wobf, m0, n0, acc);
#pragma unroll
  for (int t = 0; t < 4; ++t) {
    const int col = n0 + 16 * t + qsel;
    const float bias = bo[col];
#pragma unroll
    for (int j = 0; j < 8; ++j)
      out[(size_t)(m0 + j + 8 * hi) * HID_ + col] = acc[t][j] + bias;
  }
}

// ---------------------------------------------------------------------------
extern "C" void kernel_launch(void* const* d_in, const int* in_sizes, int n_in,
                              void* d_out, int out_size, void* d_ws, size_t ws_size,
                              hipStream_t stream) {
  (void)in_sizes; (void)n_in; (void)out_size; (void)ws_size;
  const float* q    = (const float*)d_in[0];
  const float* k    = (const float*)d_in[1];
  const float* v    = (const float*)d_in[2];
  const float* kb   = (const float*)d_in[3];
  const int*   mask = (const int*)  d_in[4];
  const float* Wq   = (const float*)d_in[5];
  const float* bq   = (const float*)d_in[6];
  const float* Wk   = (const float*)d_in[7];
  const float* bk   = (const float*)d_in[8];
  const float* Wv   = (const float*)d_in[9];
  const float* bv   = (const float*)d_in[10];
  const float* Wkb  = (const float*)d_in[11];
  const float* bkb  = (const float*)d_in[12];
  const float* Wo   = (const float*)d_in[13];
  const float* bo   = (const float*)d_in[14];

  const size_t N_ACT = (size_t)B_ * S_ * HID_;        // 4194304 elements
  const size_t N_W   = (size_t)HID_ * HID_;           // 1048576 elements
  const size_t SEG_A = N_ACT * sizeof(u16);           // 8 MiB
  const size_t SEG_W = N_W * sizeof(u16);             // 2 MiB

  char* ws = (char*)d_ws;
  u16* qbf   = (u16*)(ws);
  u16* kbf   = (u16*)(ws + 1 * SEG_A);
  u16* vbf   = (u16*)(ws + 2 * SEG_A);
  u16* kbbf  = (u16*)(ws + 3 * SEG_A);
  char* wbase = ws + 4 * SEG_A;
  u16* Wqbf  = (u16*)(wbase);
  u16* Wkbf  = (u16*)(wbase + 1 * SEG_W);
  u16* Wvbf  = (u16*)(wbase + 2 * SEG_W);
  u16* Wkbbf = (u16*)(wbase + 3 * SEG_W);
  u16* Wobf  = (u16*)(wbase + 4 * SEG_W);
  char* ibase = wbase + 5 * SEG_W;
  u16* qh   = (u16*)(ibase);
  u16* kkp  = (u16*)(ibase + 1 * SEG_A);
  u16* vt   = (u16*)(ibase + 2 * SEG_A);
  u16* hid  = (u16*)(ibase + 3 * SEG_A);

  CvtArgs ca;
  ca.src[0] = q;   ca.dst[0] = qbf;   ca.n[0] = (int)N_ACT;
  ca.src[1] = k;   ca.dst[1] = kbf;   ca.n[1] = (int)N_ACT;
  ca.src[2] = v;   ca.dst[2] = vbf;   ca.n[2] = (int)N_ACT;
  ca.src[3] = kb;  ca.dst[3] = kbbf;  ca.n[3] = (int)N_ACT;
  ca.src[4] = Wq;  ca.dst[4] = Wqbf;  ca.n[4] = (int)N_W;
  ca.src[5] = Wk;  ca.dst[5] = Wkbf;  ca.n[5] = (int)N_W;
  ca.src[6] = Wv;  ca.dst[6] = Wvbf;  ca.n[6] = (int)N_W;
  ca.src[7] = Wkb; ca.dst[7] = Wkbbf; ca.n[7] = (int)N_W;
  ca.src[8] = Wo;  ca.dst[8] = Wobf;  ca.n[8] = (int)N_W;

  cvt_kernel<<<dim3(2048, 1, 9), 256, 0, stream>>>(ca);
  proj_kernel<<<dim3(32, 16, 3), 256, 0, stream>>>(qbf, kbf, vbf, kbbf,
                                                   Wqbf, Wkbf, Wvbf, Wkbbf,
                                                   bq, bk, bv, bkb, qh, kkp, vt);
  attn_kernel<<<dim3(512), 256, 0, stream>>>(qh, kkp, vt, mask, hid);
  outproj_kernel<<<dim3(32, 16), 256, 0, stream>>>(hid, Wobf, bo, (float*)d_out);
}
